// LinearMultiheadAttention_30872224923769
// MI455X (gfx1250) — matrix operand order repaired and checked by reading, not compiled, not run on hardware
//
#include <hip/hip_runtime.h>
#include <hip/hip_bf16.h>

// ---------------- problem constants ----------------
#define L_SEQ 2048
#define NB    8
#define EMB   1024
#define HH    16
#define DH    64
#define KPC   256
#define KSZ   8
#define M1    (L_SEQ * NB)    // 16384 rows for projection GEMMs
#define M2    (KPC * NB)      // 2048 rows for compression GEMMs
#define K2    (EMB * KSZ)     // 8192 reduction dim for compression

// ---------------- WMMA types ----------------
typedef __attribute__((ext_vector_type(16))) __bf16 v16bf;
typedef __attribute__((ext_vector_type(8)))  float  v8f;

union BF16Frag {
    __bf16         b[16];
    unsigned short u[16];
    uint4          q[2];
    v16bf          v;
};

union PK8 {
    __bf16         b[8];
    unsigned short u[8];
    uint4          q;
};

__device__ __forceinline__ v8f wmma_bf16(v16bf a, v16bf b, v8f c) {
    // (neg_a, A, neg_b, B, c_mod, C, reuse_a, reuse_b)
    return __builtin_amdgcn_wmma_f32_16x16x32_bf16(false, a, false, b, (short)0, c, false, false);
}

__device__ __forceinline__ v8f zero_v8f() {
    v8f z = {0.f, 0.f, 0.f, 0.f, 0.f, 0.f, 0.f, 0.f};
    return z;
}

// native f32 -> bf16 (lowered to v_cvt_pk_bf16_f32 class ops)
__device__ __forceinline__ void cvt8_to(__bf16* d, float4 a, float4 b) {
    d[0] = (__bf16)a.x; d[1] = (__bf16)a.y; d[2] = (__bf16)a.z; d[3] = (__bf16)a.w;
    d[4] = (__bf16)b.x; d[5] = (__bf16)b.y; d[6] = (__bf16)b.z; d[7] = (__bf16)b.w;
}

// =====================================================================
// zero a float region (attn-weight output, reset every launch)
// =====================================================================
__global__ void zero_f32_kernel(float* __restrict__ p, int n) {
    int i = blockIdx.x * 256 + threadIdx.x;
    if (i < n) p[i] = 0.0f;
}

// =====================================================================
// contiguous f32 -> bf16 (8 elements/thread, b128 in / b128 out)
// =====================================================================
__global__ void f32_to_bf16_kernel(const float* __restrict__ src,
                                   __bf16* __restrict__ dst, int n8) {
    int i = blockIdx.x * 256 + threadIdx.x;
    if (i >= n8) return;
    const float4* s = (const float4*)src + (size_t)i * 2;
    PK8 pk;
    cvt8_to(pk.b, s[0], s[1]);
    *((uint4*)dst + i) = pk.q;
}

// =====================================================================
// conv-weight repack: (O, I, T) fp32 -> (O, T, I) bf16 so that
// compression-GEMM B fragments are contiguous b128 loads.
// =====================================================================
__global__ void conv_repack_kernel(const float* __restrict__ w,
                                   __bf16* __restrict__ dst) {
    int idx = blockIdx.x * 256 + threadIdx.x;        // over E*KSZ*(E/8) = 1M
    int i8   = idx & (EMB / 8 - 1);
    int rest = idx >> 7;                              // / (EMB/8)
    int t = rest & (KSZ - 1);
    int o = rest >> 3;
    const float* src = w + ((size_t)o * EMB + i8 * 8) * KSZ + t;
    PK8 pk;
    #pragma unroll
    for (int e = 0; e < 8; ++e) pk.b[e] = (__bf16)src[e * KSZ];
    *(uint4*)(dst + ((size_t)o * KSZ + t) * EMB + i8 * 8) = pk.q;
}

// =====================================================================
// Projection GEMM: Y_bf16[m,o] = (sum_k X[m,k]*W[o,k] + bias[o]) * scale
// X fp32 (converted in-register with native casts), W pre-converted bf16.
// Wave computes a 32x64 output tile (2x4 WMMA tiles).
// =====================================================================
__global__ __launch_bounds__(256) void proj_gemm_kernel(
    const float* __restrict__ X, const __bf16* __restrict__ Wbf,
    const float* __restrict__ bias, float scale,
    __bf16* __restrict__ Y)
{
    const int wid  = threadIdx.x >> 5;
    const int lane = threadIdx.x & 31;
    const int half = lane >> 4;
    const int l16  = lane & 15;

    const int ntn = EMB / 64;                 // 16 col tiles
    int tile = blockIdx.x * 8 + wid;
    int mr = (tile / ntn) * 32;
    int nc = (tile % ntn) * 64;

    v8f acc[2][4];
    #pragma unroll
    for (int i = 0; i < 2; ++i)
        #pragma unroll
        for (int j = 0; j < 4; ++j) acc[i][j] = zero_v8f();

    for (int kc = 0; kc < EMB; kc += 32) {
        BF16Frag a[2], b[4];
        #pragma unroll
        for (int i = 0; i < 2; ++i) {
            // A 16x32 bf16: lane<16 -> K {kc..kc+7, kc+16..kc+23}
            const float* arow = X + (size_t)(mr + i * 16 + l16) * EMB + kc + half * 8;
            float4 x0 = *(const float4*)(arow);
            float4 x1 = *(const float4*)(arow + 4);
            float4 x2 = *(const float4*)(arow + 16);
            float4 x3 = *(const float4*)(arow + 20);
            cvt8_to(a[i].b, x0, x1);
            cvt8_to(a[i].b + 8, x2, x3);
        }
        #pragma unroll
        for (int j = 0; j < 4; ++j) {
            // B 32x16 bf16: lane<16 -> K {kc..kc+7, kc+16..kc+23} (col = lane)
            const __bf16* bcol = Wbf + (size_t)(nc + j * 16 + l16) * EMB + kc + half * 8;
            b[j].q[0] = *(const uint4*)(bcol);
            b[j].q[1] = *(const uint4*)(bcol + 16);
        }
        #pragma unroll
        for (int i = 0; i < 2; ++i)
            #pragma unroll
            for (int j = 0; j < 4; ++j)
                acc[i][j] = wmma_bf16(a[i].v, b[j].v, acc[i][j]);
    }

    #pragma unroll
    for (int i = 0; i < 2; ++i)
        #pragma unroll
        for (int j = 0; j < 4; ++j) {
            int col = nc + j * 16 + l16;
            float bb = bias[col];
            #pragma unroll
            for (int r = 0; r < 8; ++r) {
                int m = mr + i * 16 + half * 8 + r;
                Y[(size_t)m * EMB + col] = (__bf16)((acc[i][j][r] + bb) * scale);
            }
        }
}

// =====================================================================
// Compression (conv1d k=stride=8 as GEMM, K=8192):
//   Y[(p,n),o] = sum_{t,i} Xbf[(p*8+t, n), i] * Wp[o, t, i] + bias[o]
// All-bf16 b128 fragment loads (weights pre-repacked).
// =====================================================================
__global__ __launch_bounds__(256) void compress_gemm_kernel(
    const __bf16* __restrict__ Xbf,
    const __bf16* __restrict__ Wp,
    const float* __restrict__ bias,
    __bf16* __restrict__ Y)
{
    const int wid  = threadIdx.x >> 5;
    const int lane = threadIdx.x & 31;
    const int half = lane >> 4;
    const int l16  = lane & 15;

    const int ntn = EMB / 64;                 // 16 col tiles
    int tile = blockIdx.x * 8 + wid;          // 1024 tiles total
    int mr = (tile / ntn) * 32;
    int nc = (tile % ntn) * 64;

    v8f acc[2][4];
    #pragma unroll
    for (int i = 0; i < 2; ++i)
        #pragma unroll
        for (int j = 0; j < 4; ++j) acc[i][j] = zero_v8f();

    for (int kk = 0; kk < K2; kk += 32) {
        int t  = kk >> 10;                    // conv tap
        int ib = kk & 1023;                   // input-channel base
        BF16Frag a[2], b[4];
        #pragma unroll
        for (int i = 0; i < 2; ++i) {
            int m = mr + i * 16 + l16;        // m = p*NB + n
            int p = m >> 3, n = m & 7;
            const __bf16* arow =
                Xbf + ((size_t)((p * KSZ + t) * NB + n)) * EMB + ib + half * 8;
            a[i].q[0] = *(const uint4*)(arow);
            a[i].q[1] = *(const uint4*)(arow + 16);
        }
        #pragma unroll
        for (int j = 0; j < 4; ++j) {
            int col = nc + j * 16 + l16;
            const __bf16* brow = Wp + ((size_t)col * KSZ + t) * EMB + ib + half * 8;
            b[j].q[0] = *(const uint4*)(brow);
            b[j].q[1] = *(const uint4*)(brow + 16);
        }
        #pragma unroll
        for (int i = 0; i < 2; ++i)
            #pragma unroll
            for (int j = 0; j < 4; ++j)
                acc[i][j] = wmma_bf16(a[i].v, b[j].v, acc[i][j]);
    }

    #pragma unroll
    for (int i = 0; i < 2; ++i)
        #pragma unroll
        for (int j = 0; j < 4; ++j) {
            int col = nc + j * 16 + l16;
            float bb = bias[col];
            #pragma unroll
            for (int r = 0; r < 8; ++r) {
                int m = mr + i * 16 + half * 8 + r;
                Y[(size_t)m * EMB + col] = (__bf16)(acc[i][j][r] + bb);
            }
        }
}

// =====================================================================
// Fused attention: block = 4 waves, all sharing b = blockIdx.y = n*H+h.
// Vh (256x64) is cooperatively staged *transposed* into LDS once, so
// P x V B-fragments are two ds_load_b128 each. Scores 16x256 stay in
// registers; softmax via half-wave shuffles; head-mean via f32 atomics.
// =====================================================================
__global__ __launch_bounds__(128) void attn_kernel(
    const __bf16* __restrict__ qbf,
    const __bf16* __restrict__ kcbf,
    const __bf16* __restrict__ vcbf,
    __bf16* __restrict__ obf,
    float* __restrict__ attn_out)             // (N, L, KP) fp32
{
    __shared__ __align__(16) unsigned short vT[DH][KPC];        // 32 KB
    __shared__ __align__(16) unsigned short pbuf[4][16 * KPC];  // 32 KB

    const int tid  = threadIdx.x;
    const int wid  = tid >> 5;
    const int lane = tid & 31;
    const int half = lane >> 4;
    const int l16  = lane & 15;

    const int b = blockIdx.y;                 // 0..127
    const int n = b / HH, h = b % HH;
    const int ltile = blockIdx.x * 4 + wid;   // 0..127
    const int lr = ltile * 16;

    // ---- cooperative transposed staging of Vh into LDS ----
    for (int c = tid; c < KPC * 8; c += 128) {     // (kp, d-group-of-8)
        int kp = c >> 3, dg = (c & 7) * 8;
        PK8 u;
        u.q = *(const uint4*)(vcbf + ((size_t)kp * NB + n) * EMB + h * DH + dg);
        #pragma unroll
        for (int e = 0; e < 8; ++e) vT[dg + e][kp] = u.u[e];
    }
    __syncthreads();

    // ---- Q fragments for the two K=32 chunks of D=64 ----
    BF16Frag aq0, aq1;
    {
        const __bf16* qrow = qbf + ((size_t)(lr + l16) * NB + n) * EMB + h * DH;
        aq0.q[0] = *(const uint4*)(qrow + half * 8);
        aq0.q[1] = *(const uint4*)(qrow + 16 + half * 8);
        aq1.q[0] = *(const uint4*)(qrow + 32 + half * 8);
        aq1.q[1] = *(const uint4*)(qrow + 48 + half * 8);
    }

    // ---- scores: 16 kp-tiles of 16 cols, full 16x256 in registers ----
    v8f sc[16];
    #pragma unroll
    for (int j = 0; j < 16; ++j) {
        int kp = j * 16 + l16;
        const __bf16* krow = kcbf + ((size_t)kp * NB + n) * EMB + h * DH;
        BF16Frag bk0, bk1;
        bk0.q[0] = *(const uint4*)(krow + half * 8);
        bk0.q[1] = *(const uint4*)(krow + half * 8 + 16);
        bk1.q[0] = *(const uint4*)(krow + 32 + half * 8);
        bk1.q[1] = *(const uint4*)(krow + 32 + half * 8 + 16);
        v8f c = wmma_bf16(aq0.v, bk0.v, zero_v8f());
        sc[j]  = wmma_bf16(aq1.v, bk1.v, c);
    }

    // ---- softmax over 256 cols; row r lives in 16 lanes of one half ----
    #pragma unroll
    for (int r = 0; r < 8; ++r) {
        float mx = -3.0e30f;
        #pragma unroll
        for (int j = 0; j < 16; ++j) mx = fmaxf(mx, sc[j][r]);
        mx = fmaxf(mx, __shfl_xor(mx, 1));
        mx = fmaxf(mx, __shfl_xor(mx, 2));
        mx = fmaxf(mx, __shfl_xor(mx, 4));
        mx = fmaxf(mx, __shfl_xor(mx, 8));
        float sum = 0.f;
        #pragma unroll
        for (int j = 0; j < 16; ++j) {
            float e = __expf(sc[j][r] - mx);
            sc[j][r] = e;
            sum += e;
        }
        sum += __shfl_xor(sum, 1);
        sum += __shfl_xor(sum, 2);
        sum += __shfl_xor(sum, 4);
        sum += __shfl_xor(sum, 8);
        float rinv = 1.0f / sum;
        #pragma unroll
        for (int j = 0; j < 16; ++j) sc[j][r] *= rinv;
    }

    // ---- head-mean attn weights + stage P into LDS (bf16) ----
    unsigned short* pb = pbuf[wid];
    #pragma unroll
    for (int j = 0; j < 16; ++j) {
        #pragma unroll
        for (int r = 0; r < 8; ++r) {
            int mrow = half * 8 + r;
            int kp = j * 16 + l16;
            float p = sc[j][r];
            atomicAdd(&attn_out[((size_t)n * L_SEQ + lr + mrow) * KPC + kp],
                      p * (1.0f / HH));
            union { __bf16 b; unsigned short s; } cv;
            cv.b = (__bf16)p;
            pb[mrow * KPC + kp] = cv.s;
        }
    }

    // ---- P(16x256) x Vh(256x64): both operands from LDS b128 loads ----
    #pragma unroll
    for (int dt = 0; dt < 4; ++dt) {
        v8f c = zero_v8f();
        int d = dt * 16 + l16;
        #pragma unroll
        for (int kc = 0; kc < 8; ++kc) {
            BF16Frag ap, bv;
            const unsigned short* prow = pb + l16 * KPC + kc * 32 + half * 8;
            ap.q[0] = *(const uint4*)(prow);
            ap.q[1] = *(const uint4*)(prow + 16);
            const unsigned short* vrow = &vT[d][kc * 32 + half * 8];
            bv.q[0] = *(const uint4*)(vrow);
            bv.q[1] = *(const uint4*)(vrow + 16);
            c = wmma_bf16(ap.v, bv.v, c);
        }
        #pragma unroll
        for (int r = 0; r < 8; ++r) {
            int lg = lr + half * 8 + r;
            obf[((size_t)lg * NB + n) * EMB + h * DH + dt * 16 + l16] = (__bf16)c[r];
        }
    }
}

// =====================================================================
// Output projection: d_out[m,o] = sum_k obf[m,k]*out_w[o,k] + out_b[o]
// All-bf16 fragment loads, fp32 output.
// =====================================================================
__global__ __launch_bounds__(256) void out_gemm_kernel(
    const __bf16* __restrict__ Xbf,
    const __bf16* __restrict__ Wbf,
    const float* __restrict__ bias,
    float* __restrict__ Y)
{
    const int wid  = threadIdx.x >> 5;
    const int lane = threadIdx.x & 31;
    const int half = lane >> 4;
    const int l16  = lane & 15;

    const int ntn = EMB / 64;
    int tile = blockIdx.x * 8 + wid;
    int mr = (tile / ntn) * 32;
    int nc = (tile % ntn) * 64;

    v8f acc[2][4];
    #pragma unroll
    for (int i = 0; i < 2; ++i)
        #pragma unroll
        for (int j = 0; j < 4; ++j) acc[i][j] = zero_v8f();

    for (int kc = 0; kc < EMB; kc += 32) {
        BF16Frag a[2], b[4];
        #pragma unroll
        for (int i = 0; i < 2; ++i) {
            const __bf16* arow = Xbf + (size_t)(mr + i * 16 + l16) * EMB + kc + half * 8;
            a[i].q[0] = *(const uint4*)(arow);
            a[i].q[1] = *(const uint4*)(arow + 16);
        }
        #pragma unroll
        for (int j = 0; j < 4; ++j) {
            const __bf16* bcol = Wbf + (size_t)(nc + j * 16 + l16) * EMB + kc + half * 8;
            b[j].q[0] = *(const uint4*)(bcol);
            b[j].q[1] = *(const uint4*)(bcol + 16);
        }
        #pragma unroll
        for (int i = 0; i < 2; ++i)
            #pragma unroll
            for (int j = 0; j < 4; ++j)
                acc[i][j] = wmma_bf16(a[i].v, b[j].v, acc[i][j]);
    }

    #pragma unroll
    for (int i = 0; i < 2; ++i)
        #pragma unroll
        for (int j = 0; j < 4; ++j) {
            int col = nc + j * 16 + l16;
            float bb = bias[col];
            #pragma unroll
            for (int r = 0; r < 8; ++r) {
                int m = mr + i * 16 + half * 8 + r;
                Y[(size_t)m * EMB + col] = acc[i][j][r] + bb;
            }
        }
}

// =====================================================================
extern "C" void kernel_launch(void* const* d_in, const int* in_sizes, int n_in,
                              void* d_out, int out_size, void* d_ws, size_t ws_size,
                              hipStream_t stream) {
    (void)in_sizes; (void)n_in; (void)out_size; (void)ws_size;

    const float* query     = (const float*)d_in[0];
    const float* key       = (const float*)d_in[1];
    const float* value     = (const float*)d_in[2];
    const float* in_proj_w = (const float*)d_in[3];
    const float* in_proj_b = (const float*)d_in[4];
    const float* e_w       = (const float*)d_in[5];
    const float* e_b       = (const float*)d_in[6];
    const float* f_w       = (const float*)d_in[7];
    const float* f_b       = (const float*)d_in[8];
    const float* out_w     = (const float*)d_in[9];
    const float* out_b     = (const float*)d_in[10];

    float* out      = (float*)d_out;                       // (L,N,E)
    float* attn_out = out + (size_t)L_SEQ * NB * EMB;      // (N,L,KP)

    // workspace layout (bf16 elements)
    const size_t SZ_BIG = (size_t)M1 * EMB;                // 16M
    const size_t SZ_C   = (size_t)M2 * EMB;                // 2M
    const size_t SZ_W   = (size_t)3 * EMB * EMB;           // 3M
    const size_t SZ_OW  = (size_t)EMB * EMB;               // 1M
    const size_t SZ_EW  = (size_t)EMB * EMB * KSZ;         // 8M

    __bf16* ws   = (__bf16*)d_ws;
    __bf16* qbf  = ws;
    __bf16* kbf  = ws + SZ_BIG;
    __bf16* vbf  = ws + 2 * SZ_BIG;
    __bf16* obf  = ws + 3 * SZ_BIG;
    __bf16* kcbf = ws + 4 * SZ_BIG;
    __bf16* vcbf = kcbf + SZ_C;
    __bf16* wbf  = vcbf + SZ_C;
    __bf16* owbf = wbf + SZ_W;
    __bf16* ewp  = owbf + SZ_OW;
    __bf16* fwp  = ewp + SZ_EW;

    const int attn_elems = NB * L_SEQ * KPC;               // 4,194,304
    zero_f32_kernel<<<(attn_elems + 255) / 256, 256, 0, stream>>>(attn_out, attn_elems);

    // weight preparation (bf16 conversion / conv-weight repack)
    f32_to_bf16_kernel<<<(int)(SZ_W / 8 / 256), 256, 0, stream>>>(in_proj_w, wbf,
                                                                  (int)(SZ_W / 8));
    f32_to_bf16_kernel<<<(int)(SZ_OW / 8 / 256), 256, 0, stream>>>(out_w, owbf,
                                                                   (int)(SZ_OW / 8));
    conv_repack_kernel<<<(int)(SZ_EW / 8 / 256), 256, 0, stream>>>(e_w, ewp);
    conv_repack_kernel<<<(int)(SZ_EW / 8 / 256), 256, 0, stream>>>(f_w, fwp);

    const float scaling = 0.125f;                          // D^-0.5, D=64
    // tiles: (16384/32) * (1024/64) = 8192 waves -> 1024 blocks of 8 waves
    proj_gemm_kernel<<<1024, 256, 0, stream>>>(query, wbf, in_proj_b,
                                               scaling, qbf);
    proj_gemm_kernel<<<1024, 256, 0, stream>>>(key, wbf + (size_t)EMB * EMB,
                                               in_proj_b + EMB, 1.0f, kbf);
    proj_gemm_kernel<<<1024, 256, 0, stream>>>(value, wbf + 2 * (size_t)EMB * EMB,
                                               in_proj_b + 2 * EMB, 1.0f, vbf);

    // tiles: (2048/32) * (1024/64) = 1024 waves -> 128 blocks
    compress_gemm_kernel<<<128, 256, 0, stream>>>(kbf, ewp, e_b, kcbf);
    compress_gemm_kernel<<<128, 256, 0, stream>>>(vbf, fwp, f_b, vcbf);

    attn_kernel<<<dim3(L_SEQ / 16 / 4, NB * HH), 128, 0, stream>>>(qbf, kcbf, vcbf,
                                                                   obf, attn_out);

    out_gemm_kernel<<<1024, 256, 0, stream>>>(obf, owbf, out_b, out);
}
